// NanoGPT_75720273428701
// MI455X (gfx1250) — matrix operand order repaired and checked by reading, not compiled, not run on hardware
//
#include <hip/hip_runtime.h>
#include <math.h>

typedef __attribute__((ext_vector_type(16))) _Float16 v16h;
typedef __attribute__((ext_vector_type(8)))  float    v8f;

#define NW 8            // waves per block
#define TB (NW * 32)    // threads per block
#define NB 32768        // batch
#define NT 16           // seq len
#define NC 8            // channels
#define NV 25           // vocab

__global__ void zero_loss_kernel(float* p) { *p = 0.f; }

__device__ __forceinline__ v8f wmma(v16h a, v16h b, v8f c) {
    // D = A(16x32 f16) * B(32x16 f16) + C(16x16 f32); K beyond 8/16 zero-padded
    return __builtin_amdgcn_wmma_f32_16x16x32_f16(false, a, false, b, (short)0, c, false, false);
}

// ---- DPP16 row (16-lane) reductions: VALU-only, no LDS/bpermute round-trips ----
template <int CTRL>
__device__ __forceinline__ float dppf(float x) {
    return __int_as_float(__builtin_amdgcn_update_dpp(
        0, __float_as_int(x), CTRL, 0xf, 0xf, true));
}
// row_ror:1/2/4/8 rotate-allreduce within each 16-lane row (exact for max/sum)
__device__ __forceinline__ float row_max16(float x) {
    x = fmaxf(x, dppf<0x121>(x));
    x = fmaxf(x, dppf<0x122>(x));
    x = fmaxf(x, dppf<0x124>(x));
    x = fmaxf(x, dppf<0x128>(x));
    return x;
}
__device__ __forceinline__ float row_sum16(float x) {
    x += dppf<0x121>(x);
    x += dppf<0x122>(x);
    x += dppf<0x124>(x);
    x += dppf<0x128>(x);
    return x;
}

// Pack 8 contiguous f32 (two float4 LDS b128 loads) * mask -> f16 elems 0..7, elems 8..15 zero.
__device__ __forceinline__ v16h frag_lo8(const float* src, float m) {
    const float4* p = (const float4*)src;
    float4 x0 = p[0], x1 = p[1];
    v16h a;
    a[0] = (_Float16)(x0.x * m); a[1] = (_Float16)(x0.y * m);
    a[2] = (_Float16)(x0.z * m); a[3] = (_Float16)(x0.w * m);
    a[4] = (_Float16)(x1.x * m); a[5] = (_Float16)(x1.y * m);
    a[6] = (_Float16)(x1.z * m); a[7] = (_Float16)(x1.w * m);
#pragma unroll
    for (int j = 8; j < 16; ++j) a[j] = (_Float16)0.f;
    return a;
}

// B fragment, K=8 rows x 16 cols from row-major f32 weights in LDS.
// Lanes 0-15 (m=1) hold rows 0..7 in elems 0..7; lanes 16-31 (m=0) zero.
__device__ __forceinline__ v16h frag_b8(const float* src, int rs, int co, int ln, float m) {
    v16h b;
#pragma unroll
    for (int j = 0; j < 8; ++j) b[j] = (_Float16)(src[j * rs + co + ln] * m);
#pragma unroll
    for (int j = 8; j < 16; ++j) b[j] = (_Float16)0.f;
    return b;
}

// D tile -> row-major f32 LDS. Element r is (row = r + 8*half, col = ln).
__device__ __forceinline__ void store_d(v8f d, float* dst, int rs, int co, int ln, int half) {
#pragma unroll
    for (int r = 0; r < 8; ++r) dst[(r + 8 * half) * rs + co + ln] = d[r];
}

// LayerNorm over C=8; lanes 0-15 handle one row each (xbuf row stride 16, padded),
// write normalized row to dst (row stride 32).
__device__ __forceinline__ void layernorm(const float* xw, float* dst,
                                          const float* g, const float* bta,
                                          int ln, int half) {
    if (half == 0) {
        const float4* xp = (const float4*)(xw + ln * 16);
        float4 x0 = xp[0], x1 = xp[1];
        float xr[8] = {x0.x, x0.y, x0.z, x0.w, x1.x, x1.y, x1.z, x1.w};
        float mu = 0.f;
#pragma unroll
        for (int c = 0; c < NC; ++c) mu += xr[c];
        mu *= 0.125f;
        float var = 0.f;
#pragma unroll
        for (int c = 0; c < NC; ++c) { float d = xr[c] - mu; var += d * d; }
        var *= 0.125f;
        float is = rsqrtf(var + 1e-5f);
#pragma unroll
        for (int c = 0; c < NC; ++c) dst[ln * 32 + c] = (xr[c] - mu) * is * g[c] + bta[c];
    }
}

__global__ __launch_bounds__(TB) void gpt_fwd_kernel(
    const int* __restrict__ idx, const int* __restrict__ tgt,
    const float* __restrict__ wte, const float* __restrict__ wpe,
    const float* __restrict__ ln1g, const float* __restrict__ ln1b,
    const float* __restrict__ wattn_g, const float* __restrict__ wproj_g,
    const float* __restrict__ ln2g, const float* __restrict__ ln2b,
    const float* __restrict__ wfc_g, const float* __restrict__ wmlp_g,
    const float* __restrict__ lnfg, const float* __restrict__ lnfb,
    const float* __restrict__ whead_g,
    float* __restrict__ logits, float* __restrict__ loss)
{
    __shared__ __align__(16) float wattn[8 * 32];   // qkv weights, cols 24..31 zero
    __shared__ __align__(16) float whead[8 * 32];   // head weights, cols 25..31 zero
    __shared__ __align__(16) float wproj[8 * 16];   // attn proj, cols 8..15 zero
    __shared__ __align__(16) float wfc[8 * 16];
    __shared__ __align__(16) float wmlp[8 * 16];
    __shared__ __align__(16) float lnw[6 * 8];      // ln1g ln1b ln2g ln2b lnfg lnfb
    __shared__ __align__(16) float xbuf[NW * NT * 16];  // residual, row stride 16 (padded)
    __shared__ __align__(16) float sbuf[NW * NT * 32];  // scratch per wave
    __shared__ __align__(16) float abuf[NW * NT * NT];  // attention probs per wave
    __shared__ float lred[NW];

    const int tid = threadIdx.x;
    // ---- stage weights (block-wide, once) ----
    for (int i = tid; i < 8 * 32; i += TB) {
        int r = i >> 5, c = i & 31;
        wattn[i] = (c < 24) ? wattn_g[r * 24 + c] : 0.f;
        whead[i] = (c < NV) ? whead_g[r * NV + c] : 0.f;
    }
    for (int i = tid; i < 8 * 16; i += TB) {
        int r = i >> 4, c = i & 15;
        wproj[i] = (c < 8) ? wproj_g[r * 8 + c] : 0.f;
        wfc[i]   = (c < 8) ? wfc_g[r * 8 + c]   : 0.f;
        wmlp[i]  = (c < 8) ? wmlp_g[r * 8 + c]  : 0.f;
    }
    if (tid < 8) {
        lnw[tid]      = ln1g[tid]; lnw[8 + tid]  = ln1b[tid];
        lnw[16 + tid] = ln2g[tid]; lnw[24 + tid] = ln2b[tid];
        lnw[32 + tid] = lnfg[tid]; lnw[40 + tid] = lnfb[tid];
    }
    __syncthreads();

    const int wid  = tid >> 5;
    const int lane = tid & 31;
    const int half = lane >> 4;
    const int ln   = lane & 15;
    const int b    = blockIdx.x * NW + wid;
    const float mh = (half == 0) ? 1.f : 0.f;   // arithmetic half-mask (no branches)

    float* xw = xbuf + wid * (NT * 16);
    float* sw = sbuf + wid * (NT * 32);
    float* aw = abuf + wid * (NT * NT);

    const v8f vzero = {0.f, 0.f, 0.f, 0.f, 0.f, 0.f, 0.f, 0.f};

    // ---- embedding: x = wte[idx] + wpe  (xbuf row stride 16) ----
    for (int i = lane; i < NT * NC; i += 32) {
        int t = i >> 3, c = i & 7;
        int id = idx[b * NT + t];
        xw[t * 16 + c] = wte[id * NC + c] + wpe[i];
    }

    // ---- LN1 ----
    layernorm(xw, sw, lnw + 0, lnw + 8, ln, half);

    // ---- qkv = h @ w_attn  (two 16-col tiles, K=8) ----
    {
        v16h a  = frag_lo8(sw + ln * 32, mh);
        v16h b0 = frag_b8(wattn, 32, 0,  ln, mh);
        v16h b1 = frag_b8(wattn, 32, 16, ln, mh);
        v8f q0 = wmma(a, b0, vzero);
        v8f q1 = wmma(a, b1, vzero);
        store_d(q0, sw, 32, 0,  ln, half);   // cols 0..15  (q | k)
        store_d(q1, sw, 32, 16, ln, half);   // cols 16..31 (v | pad)
    }

    // ---- att = q @ k^T / sqrt(HD), causal softmax ----
    v8f att;
    {
        v16h aq = frag_lo8(sw + ln * 32, mh);       // q rows, K=8
        v16h bk = frag_lo8(sw + ln * 32 + 8, mh);   // B[d][n] = k[n][d]: contiguous in d
        att = wmma(aq, bk, vzero);
    }
#pragma unroll
    for (int r = 0; r < 8; ++r) {
        int m = r + 8 * half;                       // query row
        float s = (ln <= m) ? att[r] * 0.3535533906f : -INFINITY;
        float mx = row_max16(s);
        float e = (ln <= m) ? __expf(s - mx) : 0.f;
        float se = row_sum16(e);
        aw[m * NT + ln] = e / se;
    }

    // ---- y = att @ v  (K=16) ----
    {
        v16h aa = frag_lo8(aw + ln * 16 + half * 8, 1.f);   // lanes 0-15: K0-7, 16-31: K8-15
        float mv = (ln < 8) ? 1.f : 0.f;
        v16h bv;                                    // B[kk][n] = v[kk][n] = sw[kk*32+16+n], kk = j + 8*half (j<8)
#pragma unroll
        for (int j = 0; j < 16; ++j) bv[j] = (j < 8) ? (_Float16)(sw[(j + 8 * half) * 32 + 16 + ln] * mv) : (_Float16)0.f;
        v8f y = wmma(aa, bv, vzero);
        // cols >= 8 are exactly zero (B zero-padded) -> store full width, branch-free
        store_d(y, sw, 32, 0, ln, half);
    }

    // ---- attn proj + residual ----
    {
        v16h a  = frag_lo8(sw + ln * 32, mh);
        v16h bp = frag_b8(wproj, 16, 0, ln, mh);
        v8f o = wmma(a, bp, vzero);
#pragma unroll
        for (int r = 0; r < 8; ++r) xw[(r + 8 * half) * 16 + ln] += o[r];  // pad cols get +0
    }

    // ---- LN2 ----
    layernorm(xw, sw, lnw + 16, lnw + 24, ln, half);

    // ---- fc + exact GELU ----
    {
        v16h a  = frag_lo8(sw + ln * 32, mh);
        v16h bf = frag_b8(wfc, 16, 0, ln, mh);
        v8f h = wmma(a, bf, vzero);
#pragma unroll
        for (int r = 0; r < 8; ++r) {
            float v = h[r];                         // cols >= 8 are exactly 0 -> gelu(0)=0
            sw[(r + 8 * half) * 32 + ln] = 0.5f * v * (1.f + erff(v * 0.7071067812f));
        }
    }

    // ---- mlp proj + residual ----
    {
        v16h a  = frag_lo8(sw + ln * 32, mh);
        v16h bm = frag_b8(wmlp, 16, 0, ln, mh);
        v8f o = wmma(a, bm, vzero);
#pragma unroll
        for (int r = 0; r < 8; ++r) xw[(r + 8 * half) * 16 + ln] += o[r];
    }

    // ---- LNf ----
    layernorm(xw, sw, lnw + 32, lnw + 40, ln, half);

    // ---- head: logits = h @ w_head  (25 cols -> two tiles) ----
    v8f l0, l1;
    {
        v16h a  = frag_lo8(sw + ln * 32, mh);
        v16h b0 = frag_b8(whead, 32, 0,  ln, mh);
        v16h b1 = frag_b8(whead, 32, 16, ln, mh);
        l0 = wmma(a, b0, vzero);
        l1 = wmma(a, b1, vzero);
    }

    // ---- store logits + cross-entropy loss ----
    float wl = 0.f;
#pragma unroll
    for (int r = 0; r < 8; ++r) {
        int t = r + 8 * half;
        float* orow = logits + ((size_t)b * NT + t) * NV;
        orow[ln] = l0[r];
        if (ln < NV - 16) orow[16 + ln] = l1[r];

        float a0 = l0[r];
        float a1 = (ln < NV - 16) ? l1[r] : -INFINITY;
        float mx = row_max16(fmaxf(a0, a1));
        float e  = __expf(a0 - mx) + ((ln < NV - 16) ? __expf(a1 - mx) : 0.f);
        e = row_sum16(e);

        // target logit: tg is uniform within the half; exactly one lane contributes
        int tg = tgt[b * NT + t];
        float pick = (tg == ln) ? a0 : ((tg == ln + 16 && ln < NV - 16) ? l1[r] : 0.f);
        float lt = row_sum16(pick);
        wl += lt - mx - __logf(e);                  // log p[target]
    }
    wl += __shfl_xor(wl, 16, 32);                   // combine the two halves (once per wave)
    if (lane == 0) lred[wid] = wl;
    __syncthreads();
    if (tid == 0) {
        float s = 0.f;
#pragma unroll
        for (int w = 0; w < NW; ++w) s += lred[w];
        atomicAdd(loss, -s * (1.0f / ((float)NB * (float)NT)));
    }
}

extern "C" void kernel_launch(void* const* d_in, const int* in_sizes, int n_in,
                              void* d_out, int out_size, void* d_ws, size_t ws_size,
                              hipStream_t stream) {
    const int*   idx   = (const int*)d_in[0];
    const int*   tgt   = (const int*)d_in[1];
    const float* wte   = (const float*)d_in[2];
    const float* wpe   = (const float*)d_in[3];
    const float* ln1g  = (const float*)d_in[4];
    const float* ln1b  = (const float*)d_in[5];
    const float* wattn = (const float*)d_in[6];
    const float* wproj = (const float*)d_in[7];
    const float* ln2g  = (const float*)d_in[8];
    const float* ln2b  = (const float*)d_in[9];
    const float* wfc   = (const float*)d_in[10];
    const float* wmlp  = (const float*)d_in[11];
    const float* lnfg  = (const float*)d_in[12];
    const float* lnfb  = (const float*)d_in[13];
    const float* whead = (const float*)d_in[14];

    float* logits = (float*)d_out;
    float* loss   = logits + (size_t)NB * NT * NV;

    zero_loss_kernel<<<1, 1, 0, stream>>>(loss);
    gpt_fwd_kernel<<<NB / NW, TB, 0, stream>>>(
        idx, tgt, wte, wpe, ln1g, ln1b, wattn, wproj,
        ln2g, ln2b, wfc, wmlp, lnfg, lnfb, whead, logits, loss);
}
